// Attention_LSTM_model_45964740001943
// MI455X (gfx1250) — compile-verified, matrix operand order checked by reading, not run on hardware
//
#include <hip/hip_runtime.h>
#include <hip/hip_bf16.h>

// ---------------------------------------------------------------------------
// Attention-LSTM for MI455X (gfx1250).  bf16 WMMA (f32 accum) for all GEMMs.
// B=128, S=512, V=50000, E=256, H=256, A=128, C=2
// ---------------------------------------------------------------------------

typedef __attribute__((ext_vector_type(16))) __bf16 v16bf;
typedef __attribute__((ext_vector_type(8)))  float  v8f;

struct __align__(16) U4 { unsigned int a, b, c, d; };
union Frag16 { v16bf v; U4 q[2]; };

__device__ __forceinline__ unsigned short f2bf(float f) {
  union { float f; unsigned int u; } v; v.f = f;
  unsigned int r = v.u + 0x7FFFu + ((v.u >> 16) & 1u);   // round-to-nearest-even
  return (unsigned short)(r >> 16);
}
__device__ __forceinline__ float bf2f(unsigned short h) {
  union { unsigned int u; float f; } v; v.u = ((unsigned int)h) << 16;
  return v.f;
}

// Load one 16x32 bf16 WMMA operand fragment from a row-major matrix.
// Per ISA 7.12.2: lane L holds row/col (L&15); lanes 16-31 are K+8.
// Each lane's 16 elements are two contiguous 16B chunks at K = base and base+16.
__device__ __forceinline__ v16bf load_frag(const unsigned short* mat, int ld,
                                           int row_base, int k_base) {
  int lane = threadIdx.x & 31;
  int r = row_base + (lane & 15);
  int k = k_base + ((lane >> 4) << 3);
  const U4* p = reinterpret_cast<const U4*>(mat + (size_t)r * ld + k);
  Frag16 f;
  f.q[0] = p[0];   // K rel +0..+7
  f.q[1] = p[2];   // K rel +16..+23   (+32 bytes)
  return f.v;
}

__device__ __forceinline__ v8f wmma_bf16(v16bf a, v16bf b, v8f c) {
  return __builtin_amdgcn_wmma_f32_16x16x32_bf16(false, a, false, b,
                                                 (short)0, c, false, false);
}

// ---------------------------------------------------------------------------
// Prep kernels
// ---------------------------------------------------------------------------
__global__ void convert_kernel(const float* __restrict__ src,
                               unsigned short* __restrict__ dst, int n) {
  int i = blockIdx.x * blockDim.x + threadIdx.x;
  if (i < n) dst[i] = f2bf(src[i]);
}

__global__ void bias_kernel(const float* __restrict__ a,
                            const float* __restrict__ b,
                            float* __restrict__ o, int n) {
  int i = blockIdx.x * blockDim.x + threadIdx.x;
  if (i < n) o[i] = a[i] + b[i];
}

// embB[(b*S+s)*E + e] = bf16( embedding[x[b,s]*E + e] )
__global__ void gather_kernel(const int* __restrict__ x,
                              const float* __restrict__ emb,
                              unsigned short* __restrict__ embB) {
  size_t gid = (size_t)blockIdx.x * blockDim.x + threadIdx.x;  // B*S*E threads
  size_t tok = gid >> 8;          // / E (=256)
  int    e   = (int)(gid & 255);
  int    row = x[tok];
  embB[gid] = f2bf(emb[(size_t)row * 256 + e]);
}

// ---------------------------------------------------------------------------
// x_proj[M=65536, N=1024] = embB(bf16) @ W_ih^T(bf16) + bias   (f32 out)
// One wave computes a 16x64 tile (A reused over 4 column tiles).
// ---------------------------------------------------------------------------
__global__ __launch_bounds__(256) void xproj_kernel(
    const unsigned short* __restrict__ embB,
    const unsigned short* __restrict__ WihB,
    const float* __restrict__ bias,
    float* __restrict__ xproj) {
  int wid  = blockIdx.x * 8 + (threadIdx.x >> 5);
  int lane = threadIdx.x & 31;
  int tm = wid >> 4;              // 4096 row tiles of 16
  int tn = wid & 15;              // 16 column blocks of 64
  int rbase = tm * 16;
  v8f acc[4] = {};
  for (int kt = 0; kt < 8; ++kt) {
    v16bf a = load_frag(embB, 256, rbase, kt * 32);
#pragma unroll
    for (int ct = 0; ct < 4; ++ct) {
      v16bf b = load_frag(WihB, 256, tn * 64 + ct * 16, kt * 32);
      acc[ct] = wmma_bf16(a, b, acc[ct]);
    }
  }
  int n = lane & 15, hi = lane >> 4;
#pragma unroll
  for (int ct = 0; ct < 4; ++ct) {
    int col = tn * 64 + ct * 16 + n;
    float bv = bias[col];
#pragma unroll
    for (int r = 0; r < 8; ++r) {
      int row = rbase + r + hi * 8;
      xproj[(size_t)row * 1024 + col] = acc[ct][r] + bv;
    }
  }
}

// ---------------------------------------------------------------------------
// Persistent LSTM. Grid = 8 blocks (one 16-row batch tile each), 16 waves.
// Wave w owns hidden columns [w*16, w*16+16) for ALL four gates -> the cell
// update is wave-local. h_t lives in an 8KB LDS bf16 tile shared as the
// WMMA A operand for the next step. c stays in registers.
// ---------------------------------------------------------------------------
__global__ __launch_bounds__(512) void lstm_kernel(
    const float* __restrict__ xproj,
    const unsigned short* __restrict__ WhhB,
    unsigned short* __restrict__ hall) {
  __shared__ __align__(16) unsigned short hB[16 * 256];   // h_{t-1}, bf16
  int tid  = threadIdx.x;
  int w    = tid >> 5;
  int lane = tid & 31;
  int n = lane & 15, hi = lane >> 4;
  int bt = blockIdx.x;            // batch tile (16 rows)
  int jb = w * 16;                // this wave's hidden-column base

  for (int i = tid; i < 16 * 256; i += 512) hB[i] = 0;
  __syncthreads();

  v8f cReg = {};                  // cell state for (row r+hi*8, col jb+n)

  for (int t = 0; t < 512; ++t) {
    // Initialize accumulators with the precomputed input projection.
    v8f acc[4];
#pragma unroll
    for (int g = 0; g < 4; ++g) {
      int col = g * 256 + jb + n;
#pragma unroll
      for (int r = 0; r < 8; ++r) {
        size_t row = (size_t)(bt * 16 + r + hi * 8) * 512 + t;
        acc[g][r] = xproj[row * 1024 + col];
      }
    }
    // gates += h_{t-1} @ W_hh^T   (K = 256 -> 8 WMMA k-steps)
#pragma unroll
    for (int kt = 0; kt < 8; ++kt) {
      v16bf a = load_frag(hB, 256, 0, kt * 32);
#pragma unroll
      for (int g = 0; g < 4; ++g) {
        v16bf b = load_frag(WhhB, 256, g * 256 + jb, kt * 32);
        acc[g] = wmma_bf16(a, b, acc[g]);
      }
    }
    __syncthreads();              // everyone finished reading h_{t-1}
#pragma unroll
    for (int r = 0; r < 8; ++r) {
      float ig = 1.f / (1.f + __expf(-acc[0][r]));
      float fg = 1.f / (1.f + __expf(-acc[1][r]));
      float gg = tanhf(acc[2][r]);
      float og = 1.f / (1.f + __expf(-acc[3][r]));
      float c  = fg * cReg[r] + ig * gg;
      cReg[r]  = c;
      float h  = og * tanhf(c);
      int m = r + hi * 8;
      unsigned short hb = f2bf(h);
      hB[m * 256 + jb + n] = hb;
      hall[((size_t)(bt * 16 + m) * 512 + t) * 256 + jb + n] = hb;
    }
    __syncthreads();              // h_t visible before next step's loads
  }
}

// ---------------------------------------------------------------------------
// scores[row] = sum_a tanh( (h @ W1^T)[row,a] + b1[a] ) * U[a]
// One wave per 16 rows; 8 column tiles (A=128); shuffle-reduce over lanes.
// ---------------------------------------------------------------------------
__global__ __launch_bounds__(256) void score_kernel(
    const unsigned short* __restrict__ hall,
    const unsigned short* __restrict__ W1B,
    const float* __restrict__ b1,
    const float* __restrict__ U,
    float* __restrict__ scores) {
  int wid  = blockIdx.x * 8 + (threadIdx.x >> 5);
  int lane = threadIdx.x & 31;
  int rbase = wid * 16;
  int n = lane & 15, hi = lane >> 4;
  v8f acc[8] = {};
  for (int kt = 0; kt < 8; ++kt) {
    v16bf a = load_frag(hall, 256, rbase, kt * 32);
#pragma unroll
    for (int ct = 0; ct < 8; ++ct) {
      v16bf b = load_frag(W1B, 256, ct * 16, kt * 32);
      acc[ct] = wmma_bf16(a, b, acc[ct]);
    }
  }
#pragma unroll
  for (int r = 0; r < 8; ++r) {
    float s = 0.f;
#pragma unroll
    for (int ct = 0; ct < 8; ++ct) {
      int col = ct * 16 + n;
      s += tanhf(acc[ct][r] + b1[col]) * U[col];
    }
    s += __shfl_xor(s, 1, 16);
    s += __shfl_xor(s, 2, 16);
    s += __shfl_xor(s, 4, 16);
    s += __shfl_xor(s, 8, 16);
    if (n == 0) scores[rbase + r + hi * 8] = s;
  }
}

// ---------------------------------------------------------------------------
// Row softmax over S=512 (one block per batch row)
// ---------------------------------------------------------------------------
__global__ __launch_bounds__(512) void softmax_kernel(
    const float* __restrict__ scores, float* __restrict__ alpha) {
  __shared__ float red[512];
  int b = blockIdx.x, s = threadIdx.x;
  float v = scores[b * 512 + s];
  red[s] = v; __syncthreads();
  for (int off = 256; off > 0; off >>= 1) {
    if (s < off) red[s] = fmaxf(red[s], red[s + off]);
    __syncthreads();
  }
  float m = red[0]; __syncthreads();
  float e = __expf(v - m);
  red[s] = e; __syncthreads();
  for (int off = 256; off > 0; off >>= 1) {
    if (s < off) red[s] += red[s + off];
    __syncthreads();
  }
  alpha[b * 512 + s] = e / red[0];
}

// ctx[b,j] = sum_s alpha[b,s] * h[b,s,j]
__global__ __launch_bounds__(256) void ctx_kernel(
    const float* __restrict__ alpha,
    const unsigned short* __restrict__ hall,
    float* __restrict__ ctx) {
  int b = blockIdx.x, j = threadIdx.x;
  float acc = 0.f;
  for (int s = 0; s < 512; ++s)
    acc += alpha[b * 512 + s] * bf2f(hall[((size_t)b * 512 + s) * 256 + j]);
  ctx[b * 256 + j] = acc;
}

// logit[b,c] = ctx[b,:] . W2[c,:] + b2[c]   (128x2, tiny)
__global__ __launch_bounds__(256) void logit_kernel(
    const float* __restrict__ ctx, const float* __restrict__ W2,
    const float* __restrict__ b2, float* __restrict__ out) {
  int o = threadIdx.x;            // 256 outputs = 128*2
  int b = o >> 1, c = o & 1;
  float s = b2[c];
  for (int j = 0; j < 256; ++j) s += ctx[b * 256 + j] * W2[c * 256 + j];
  out[b * 2 + c] = s;
}

// ---------------------------------------------------------------------------
// Workspace layout (256B aligned)
// ---------------------------------------------------------------------------
static constexpr size_t OFF_WIHB   = 0;           //  1024*256 bf16
static constexpr size_t OFF_WHHB   = 524288;      //  1024*256 bf16
static constexpr size_t OFF_W1B    = 1048576;     //   128*256 bf16
static constexpr size_t OFF_BIAS   = 1114112;     //  1024 f32
static constexpr size_t OFF_EMBB   = 1118208;     // 65536*256 bf16
static constexpr size_t OFF_XPROJ  = 34672640;    // 65536*1024 f32
static constexpr size_t OFF_HALL   = 303108096;   // 65536*256 bf16
static constexpr size_t OFF_SCORES = 336662528;   // 65536 f32
static constexpr size_t OFF_ALPHA  = 336924672;   // 65536 f32
static constexpr size_t OFF_CTX    = 337186816;   // 128*256 f32

extern "C" void kernel_launch(void* const* d_in, const int* in_sizes, int n_in,
                              void* d_out, int out_size, void* d_ws, size_t ws_size,
                              hipStream_t stream) {
  const int*   x   = (const int*)  d_in[0];
  const float* emb = (const float*)d_in[1];
  const float* Wih = (const float*)d_in[2];
  const float* Whh = (const float*)d_in[3];
  const float* bih = (const float*)d_in[4];
  const float* bhh = (const float*)d_in[5];
  const float* W1  = (const float*)d_in[6];
  const float* b1  = (const float*)d_in[7];
  const float* U   = (const float*)d_in[8];
  const float* W2  = (const float*)d_in[9];
  const float* b2  = (const float*)d_in[10];
  float* out = (float*)d_out;

  char* ws = (char*)d_ws;
  unsigned short* WihB  = (unsigned short*)(ws + OFF_WIHB);
  unsigned short* WhhB  = (unsigned short*)(ws + OFF_WHHB);
  unsigned short* W1B   = (unsigned short*)(ws + OFF_W1B);
  float*          bias  = (float*)         (ws + OFF_BIAS);
  unsigned short* embB  = (unsigned short*)(ws + OFF_EMBB);
  float*          xproj = (float*)         (ws + OFF_XPROJ);
  unsigned short* hall  = (unsigned short*)(ws + OFF_HALL);
  float*          score = (float*)         (ws + OFF_SCORES);
  float*          alpha = (float*)         (ws + OFF_ALPHA);
  float*          ctx   = (float*)         (ws + OFF_CTX);

  convert_kernel<<<1024, 256, 0, stream>>>(Wih, WihB, 1024 * 256);
  convert_kernel<<<1024, 256, 0, stream>>>(Whh, WhhB, 1024 * 256);
  convert_kernel<<<128,  256, 0, stream>>>(W1,  W1B,  128 * 256);
  bias_kernel<<<4, 256, 0, stream>>>(bih, bhh, bias, 1024);

  gather_kernel<<<65536, 256, 0, stream>>>(x, emb, embB);        // B*S*E threads
  xproj_kernel <<<8192, 256, 0, stream>>>(embB, WihB, bias, xproj);
  lstm_kernel  <<<8,    512, 0, stream>>>(xproj, WhhB, hall);
  score_kernel <<<512,  256, 0, stream>>>(hall, W1B, b1, U, score);
  softmax_kernel<<<128, 512, 0, stream>>>(score, alpha);
  ctx_kernel   <<<128,  256, 0, stream>>>(alpha, hall, ctx);
  logit_kernel <<<1,    256, 0, stream>>>(ctx, W2, b2, out);
}
